// DRCNN_62165356642340
// MI455X (gfx1250) — matrix operand order repaired and checked by reading, not compiled, not run on hardware
//
#include <hip/hip_runtime.h>
#include <hip/hip_bf16.h>

typedef __attribute__((ext_vector_type(16))) _Float16 v16h;
typedef __attribute__((ext_vector_type(8)))  _Float16 v8h;
typedef __attribute__((ext_vector_type(8)))  float    v8f;

#define IMG_H 375
#define IMG_W 1242
#define MAXC  96
#define ST    104   // LDS row stride in halfs (96 + 8 pad); 208 B = 13*16 -> 16B-aligned rows

// Build a v16h WMMA fragment from two 16-byte LDS chunks.
__device__ inline v16h ld_frag(const _Float16* c0, const _Float16* c1) {
    v8h lo = *(const v8h*)c0;
    v8h hi = *(const v8h*)c1;
    return __builtin_shufflevector(lo, hi, 0,1,2,3,4,5,6,7,8,9,10,11,12,13,14,15);
}

__global__ __launch_bounds__(256)
void ssim_pair_kernel(const float* __restrict__ imgL,
                      const float* __restrict__ imgR,
                      const int*   __restrict__ lboxes,
                      const int*   __restrict__ rboxes,
                      float*       __restrict__ out)
{
    __shared__ __align__(16) _Float16 sL[MAXC * ST];
    __shared__ __align__(16) _Float16 sR[MAXC * ST];
    __shared__ __align__(16) _Float16 sB[MAXC * ST];      // dense symmetric gaussian band matrix
    __shared__ __align__(16) _Float16 sHT[5][MAXC * ST];  // horizontal-pass results, TRANSPOSED [x][y]
    __shared__ float sg[16];
    __shared__ float sPart[256];

    const int p    = blockIdx.x;          // pair index, row-major (NL, NR)
    const int bi   = p >> 5;
    const int bj   = p & 31;
    const int tid  = threadIdx.x;
    const int lane = tid & 31;
    const int w    = tid >> 5;            // wave id 0..7 (uniform per wave)
    const int L    = lane & 15;
    const int hi   = lane >> 4;

    // ---- per-pair box scalars ----
    const int x1  = lboxes[bi*4+0], y1  = lboxes[bi*4+1];
    const int x2  = lboxes[bi*4+2], y2  = lboxes[bi*4+3];
    const int x1p = rboxes[bj*4+0], y1p = rboxes[bj*4+1];
    const int x2p = rboxes[bj*4+2], y2p = rboxes[bj*4+3];

    const float hmean = 0.5f * (float)((y2 - y1) + (y2p - y1p));
    const float cd    = 0.5f * (float)((x1 + x2) - (x1p + x2p));
    const bool  gate  = fabsf(cd - (hmean * 0.2f + 10.0f)) < 120.0f;

    int wp = max(x2 - x1, x2p - x1p);
    wp = min(wp, min(IMG_W - x1, IMG_W - x1p));
    int hp = max(y2 - y1, y2p - y1p);
    hp = min(hp, min(IMG_H - y1, IMG_H - y1p));

    // ---- normalized 11-tap gaussian ----
    if (tid == 0) {
        float tmp[11]; float s = 0.0f;
        #pragma unroll
        for (int i = 0; i < 11; ++i) {
            float x = (float)(i - 5);
            tmp[i] = __expf(-x * x / 4.5f);
            s += tmp[i];
        }
        float inv = 1.0f / s;
        #pragma unroll
        for (int i = 0; i < 11; ++i) sg[i] = tmp[i] * inv;
    }
    __syncthreads();

    // ---- materialize dense banded gaussian matrix once: B[r][c] = g[c-r+5] ----
    for (int idx = tid; idx < MAXC * MAXC; idx += 256) {
        int r = idx / MAXC, c = idx - r * MAXC;
        int d = c - r + 5;
        float v = ((unsigned)d <= 10u) ? sg[d] : 0.0f;
        sB[r * ST + c] = (_Float16)v;
    }

    float thrSum = 0.0f;
    const float inv255 = 1.0f / 255.0f;

    for (int c = 0; c < 3; ++c) {
        __syncthreads();   // protect sL/sR/sHT from previous iteration readers
        // ---- stage masked crops into LDS as f16 ----
        for (int px = tid; px < MAXC * MAXC; px += 256) {
            int y = px / MAXC, x = px - y * MAXC;
            float m  = (y < hp && x < wp) ? inv255 : 0.0f;
            float vl = imgL[(((y1  + y) * IMG_W) + (x1  + x)) * 3 + c] * m;
            float vr = imgR[(((y1p + y) * IMG_W) + (x1p + x)) * 3 + c] * m;
            sL[y * ST + x] = (_Float16)vl;
            sR[y * ST + x] = (_Float16)vr;
        }
        __syncthreads();

        // ---- horizontal pass: H_q = X_q * B ; store transposed HT[x][y] ----
        for (int t = w; t < 36; t += 8) {
            int i = t / 6, j = t % 6;
            v8f z = {};
            v8f a0 = z, a1 = z, a2 = z, a3 = z, a4 = z;
            #pragma unroll
            for (int kb = 0; kb < 3; ++kb) {
                const int m = 16 * i + L;
                const int n = 16 * j + L;
                // A fragments: rows of X, chunks at k0+16*hi and k0+16*hi+8
                const _Float16* rl = &sL[m * ST + 32 * kb];
                const _Float16* rr = &sR[m * ST + 32 * kb];
                v16h aL = ld_frag(rl + 16 * hi, rl + 16 * hi + 8);
                v16h aR = ld_frag(rr + 16 * hi, rr + 16 * hi + 8);
                v16h aLL = aL * aL;
                v16h aRR = aR * aR;
                v16h aLR = aL * aR;
                // B fragment: B[k][n] = B[n][k] (symmetric) -> contiguous k along row n
                const _Float16* rb = &sB[n * ST + 32 * kb + 16 * hi];
                v16h bB = ld_frag(rb, rb + 8);
                a0 = __builtin_amdgcn_wmma_f32_16x16x32_f16(false, aL,  false, bB, (short)0, a0, false, false);
                a1 = __builtin_amdgcn_wmma_f32_16x16x32_f16(false, aR,  false, bB, (short)0, a1, false, false);
                a2 = __builtin_amdgcn_wmma_f32_16x16x32_f16(false, aLL, false, bB, (short)0, a2, false, false);
                a3 = __builtin_amdgcn_wmma_f32_16x16x32_f16(false, aRR, false, bB, (short)0, a3, false, false);
                a4 = __builtin_amdgcn_wmma_f32_16x16x32_f16(false, aLR, false, bB, (short)0, a4, false, false);
            }
            // C rows r are consecutive y at fixed x -> one b128 store per quantity (transposed)
            {
                const int x = 16 * j + L;
                const int y0 = 16 * i + 8 * hi;
                v8h h0, h1, h2, h3, h4;
                #pragma unroll
                for (int r = 0; r < 8; ++r) {
                    h0[r] = (_Float16)a0[r];
                    h1[r] = (_Float16)a1[r];
                    h2[r] = (_Float16)a2[r];
                    h3[r] = (_Float16)a3[r];
                    h4[r] = (_Float16)a4[r];
                }
                const int off = x * ST + y0;
                *(v8h*)&sHT[0][off] = h0;
                *(v8h*)&sHT[1][off] = h1;
                *(v8h*)&sHT[2][off] = h2;
                *(v8h*)&sHT[3][off] = h3;
                *(v8h*)&sHT[4][off] = h4;
            }
        }
        __syncthreads();

        // ---- vertical pass: out_q = B * H_q, fused SSIM + masked reduce ----
        for (int t = w; t < 36; t += 8) {
            int i = t / 6, j = t % 6;
            v8f z = {};
            v8f a0 = z, a1 = z, a2 = z, a3 = z, a4 = z;
            #pragma unroll
            for (int kb = 0; kb < 3; ++kb) {
                const int m = 16 * i + L;
                const int n = 16 * j + L;
                // A fragment: band matrix rows, chunks at k0+16*hi and k0+16*hi+8
                const _Float16* rbm = &sB[m * ST + 32 * kb];
                v16h aB = ld_frag(rbm + 16 * hi, rbm + 16 * hi + 8);
                // B fragments: H[k][n] = HT[n][k] -> contiguous k along HT row n
                const int off = n * ST + 32 * kb + 16 * hi;
                v16h b0 = ld_frag(&sHT[0][off], &sHT[0][off] + 8);
                v16h b1 = ld_frag(&sHT[1][off], &sHT[1][off] + 8);
                v16h b2 = ld_frag(&sHT[2][off], &sHT[2][off] + 8);
                v16h b3 = ld_frag(&sHT[3][off], &sHT[3][off] + 8);
                v16h b4 = ld_frag(&sHT[4][off], &sHT[4][off] + 8);
                a0 = __builtin_amdgcn_wmma_f32_16x16x32_f16(false, aB, false, b0, (short)0, a0, false, false);
                a1 = __builtin_amdgcn_wmma_f32_16x16x32_f16(false, aB, false, b1, (short)0, a1, false, false);
                a2 = __builtin_amdgcn_wmma_f32_16x16x32_f16(false, aB, false, b2, (short)0, a2, false, false);
                a3 = __builtin_amdgcn_wmma_f32_16x16x32_f16(false, aB, false, b3, (short)0, a3, false, false);
                a4 = __builtin_amdgcn_wmma_f32_16x16x32_f16(false, aB, false, b4, (short)0, a4, false, false);
            }
            #pragma unroll
            for (int r = 0; r < 8; ++r) {
                int y = 16 * i + r + 8 * hi;
                int x = 16 * j + L;
                if (y < hp && x < wp) {
                    float mu1 = a0[r], mu2 = a1[r];
                    float mu1s = mu1 * mu1, mu2s = mu2 * mu2, mu12 = mu1 * mu2;
                    float s1  = a2[r] - mu1s;
                    float s2  = a3[r] - mu2s;
                    float s12 = a4[r] - mu12;
                    float num = (2.0f * mu12 + 1e-4f) * (2.0f * s12 + 9e-4f);
                    float den = (mu1s + mu2s + 1e-4f) * (s1 + s2 + 9e-4f);
                    thrSum += num / den;
                }
            }
        }
    }
    __syncthreads();

    // ---- block reduction + final score ----
    sPart[tid] = thrSum;
    __syncthreads();
    #pragma unroll
    for (int s = 128; s > 0; s >>= 1) {
        if (tid < s) sPart[tid] += sPart[tid + s];
        __syncthreads();
    }
    if (tid == 0) {
        float denom = 3.0f * (float)hp * (float)wp;
        out[p] = gate ? (sPart[0] / denom) : -10.0f;
    }
}

// Greedy global-argmax matching over the 32x32 score matrix (in d_out[0..1023]).
__global__ __launch_bounds__(1024)
void greedy_match_kernel(float* __restrict__ out)
{
    __shared__ float S[1024];
    __shared__ float mv[1024];
    __shared__ int   mi[1024];
    const int t = threadIdx.x;
    S[t] = out[t];
    __syncthreads();

    for (int it = 0; it < 32; ++it) {
        mv[t] = S[t];
        mi[t] = t;
        __syncthreads();
        for (int s = 512; s > 0; s >>= 1) {
            if (t < s) {
                float v2 = mv[t + s]; int i2 = mi[t + s];
                // jnp.argmax tie-break: lowest flat index wins
                if (v2 > mv[t] || (v2 == mv[t] && i2 < mi[t])) { mv[t] = v2; mi[t] = i2; }
            }
            __syncthreads();
        }
        int   best = mi[0];
        float bv   = mv[0];
        int   row  = best >> 5, col = best & 31;
        if (t == 0) {
            out[1024 + it] = (float)row;
            out[1056 + it] = (float)col;
            out[1088 + it] = (bv > 0.0f) ? 1.0f : 0.0f;
        }
        __syncthreads();
        if ((t >> 5) == row || (t & 31) == col) S[t] = fminf(S[t], 0.0f);
        __syncthreads();
    }
}

extern "C" void kernel_launch(void* const* d_in, const int* in_sizes, int n_in,
                              void* d_out, int out_size, void* d_ws, size_t ws_size,
                              hipStream_t stream)
{
    const float* imgL   = (const float*)d_in[0];   // (375,1242,3) f32
    const float* imgR   = (const float*)d_in[1];
    const int*   lboxes = (const int*)d_in[2];     // (32,4) i32
    const int*   rboxes = (const int*)d_in[3];     // (32,4) i32
    float*       out    = (float*)d_out;           // 1024 ssims + 32 lidx + 32 ridx + 32 valid

    ssim_pair_kernel<<<dim3(1024), dim3(256), 0, stream>>>(imgL, imgR, lboxes, rboxes, out);
    greedy_match_kernel<<<dim3(1), dim3(1024), 0, stream>>>(out);
}
